// my_IAICD_90769838834050
// MI455X (gfx1250) — matrix operand order repaired and checked by reading, not compiled, not run on hardware
//
#include <hip/hip_runtime.h>
#include <hip/hip_bf16.h>

#define BB 4
#define CC 64
#define HH 160
#define WW 320
#define NPIX (BB*CC*HH*WW)

typedef __bf16 bf16x16 __attribute__((ext_vector_type(16)));
typedef float  f32x8   __attribute__((ext_vector_type(8)));

union FragU { bf16x16 v; unsigned int u[8]; };

__device__ __forceinline__ unsigned short f2bf(float f) {
    union { float f; unsigned int u; } x; x.f = f;
    unsigned int r = x.u + 0x7FFFu + ((x.u >> 16) & 1u);  // round-to-nearest-even
    return (unsigned short)(r >> 16);
}

// ---------------------------------------------------------------------------
// Unified implicit-GEMM conv kernel (3x3 when T==9, 1x1 when T==1), bf16 WMMA.
// Compile-time specialized on (CIN, T, MODE) so the tap x k-chunk loops fully
// unroll into T*(CIN/32)*4 static v_wmma_f32_16x16x32_bf16 ops and the
// epilogue has no branches.
// Block: 128 threads (4 waves). Block tile: M=64 (all out channels) x N=64
// pixels along one row. Wave tile: M=64 x N=16 -> 4 f32 16x16 accumulators.
// MODE epilogues:
//   0: outf = acc
//   1: y = acc*p0[oc] + p1[oc]; relu; outf = y; outbf = bf16(y)   (BN+ReLU)
//   2: y = p0[idx] - p1[0]*acc; outbf = bf16(y)                   (residual)
//   3: outf = acc + p0[oc]                                        (bias)
// ---------------------------------------------------------------------------
template <int CIN, int T, int MODE>
__global__ __launch_bounds__(128) void conv_wmma_kernel(
    const unsigned short* __restrict__ in_bf,
    const unsigned int* __restrict__ wfrag,
    const float* __restrict__ p0, const float* __restrict__ p1,
    float* __restrict__ outf, unsigned short* __restrict__ outbf, int obf_cs)
{
    constexpr int KC = CIN / 32;
    __shared__ unsigned short tile[3 * CIN * 66];
    const int w0  = blockIdx.x * 64;
    const int h   = blockIdx.y;
    const int b   = blockIdx.z;
    const int tid = threadIdx.x;

    // Stage 3-row halo tile (zero-padded) as bf16 into LDS.
    constexpr int TOTAL = 3 * CIN * 66;
    for (int i = tid; i < TOTAL; i += 128) {
        int r   = i / (CIN * 66);
        int rem = i - r * (CIN * 66);
        int ci  = rem / 66;
        int wl  = rem - ci * 66;
        int ih  = h + r - 1;
        int iw  = w0 + wl - 1;
        unsigned short val = 0;
        if (ih >= 0 && ih < HH && iw >= 0 && iw < WW)
            val = in_bf[(((size_t)b * CIN + ci) * HH + ih) * WW + iw];
        tile[(r * CIN + ci) * 66 + wl] = val;
    }
    __syncthreads();

    const int wid  = tid >> 5;
    const int lane = tid & 31;
    const int n    = lane & 15;
    const int half = lane >> 4;

    f32x8 acc[4];
    #pragma unroll
    for (int mt = 0; mt < 4; ++mt)
        #pragma unroll
        for (int e = 0; e < 8; ++e) acc[mt][e] = 0.f;

    #pragma unroll
    for (int t = 0; t < T; ++t) {
        constexpr bool is3x3 = (T == 9);
        const int dy = is3x3 ? (t / 3) : 1;
        const int dx = is3x3 ? (t % 3) : 1;
        const int wl = wid * 16 + n + dx;   // pixel p -> LDS col p+1, tap offset dx-1
        // prefetch next tap's weight fragments while this tap computes
        if (t + 1 < T)
            __builtin_prefetch(wfrag + ((((size_t)(t + 1) * KC) * 32 + lane) << 3), 0, 3);
        #pragma unroll
        for (int kc = 0; kc < KC; ++kc) {
            // B fragment (32x16 K-major): lane layout K = j<4 ? 2j+8*half : 16+2(j-4)+8*half (+0/1)
            FragU bfr;
            #pragma unroll
            for (int j = 0; j < 8; ++j) {
                int ci = kc * 32 + ((j < 4) ? (2 * j + 8 * half) : (16 + 2 * (j - 4) + 8 * half));
                unsigned int e0 = tile[(dy * CIN + ci    ) * 66 + wl];
                unsigned int e1 = tile[(dy * CIN + ci + 1) * 66 + wl];
                bfr.u[j] = e0 | (e1 << 16);
            }
            // A fragments: pre-swizzled weights, 8 contiguous dwords per lane.
            #pragma unroll
            for (int mt = 0; mt < 4; ++mt) {
                FragU afr;
                const unsigned int* wp =
                    wfrag + ((((size_t)(mt * T + t) * KC + kc) * 32 + lane) << 3);
                #pragma unroll
                for (int j = 0; j < 8; ++j) afr.u[j] = wp[j];
                acc[mt] = __builtin_amdgcn_wmma_f32_16x16x32_bf16(
                    false, afr.v, false, bfr.v, (short)0, acc[mt], false, false);
            }
        }
    }

    // D layout: VGPR v -> M = v + 8*half, N = lane%16.
    const int w = w0 + wid * 16 + n;
    #pragma unroll
    for (int mt = 0; mt < 4; ++mt) {
        #pragma unroll
        for (int vv = 0; vv < 8; ++vv) {
            int oc = mt * 16 + vv + 8 * half;
            float a = acc[mt][vv];
            size_t oidx = (((size_t)b * CC + oc) * HH + h) * WW + w;
            if constexpr (MODE == 0) {
                outf[oidx] = a;
            } else if constexpr (MODE == 1) {
                float y = fmaxf(a * p0[oc] + p1[oc], 0.f);
                outf[oidx] = y;
                outbf[(((size_t)b * obf_cs + oc) * HH + h) * WW + w] = f2bf(y);
            } else if constexpr (MODE == 2) {
                float y = p0[oidx] - p1[0] * a;
                outbf[(((size_t)b * obf_cs + oc) * HH + h) * WW + w] = f2bf(y);
            } else {
                outf[oidx] = a + p0[oc];
            }
        }
    }
}

// Pre-swizzle a weight tensor W[64][Cin][T] into CDNA5 A-matrix fragments:
// frag[(((mt*T+t)*KC+kc)*32+lane)*8+j], A K-pattern: j<4 -> 2j+8h, else 16+2(j-4)+8h.
__global__ void frag_weights_kernel(const float* __restrict__ Wt, int Cin, int T,
                                    unsigned int* __restrict__ frag, int total)
{
    int KC = Cin >> 5;
    for (int i = blockIdx.x * blockDim.x + threadIdx.x; i < total;
         i += gridDim.x * blockDim.x) {
        int j    = i & 7;
        int lane = (i >> 3) & 31;
        int rest = i >> 8;
        int kc   = rest % KC;
        int t    = (rest / KC) % T;
        int mt   = rest / (KC * T);
        int m    = mt * 16 + (lane & 15);
        int hv   = lane >> 4;
        int kk   = (j < 4) ? (2 * j + 8 * hv) : (16 + 2 * (j - 4) + 8 * hv);
        int ci   = kc * 32 + kk;
        unsigned int e0 = f2bf(Wt[((size_t)m * Cin + ci    ) * T + t]);
        unsigned int e1 = f2bf(Wt[((size_t)m * Cin + ci + 1) * T + t]);
        frag[i] = e0 | (e1 << 16);
    }
}

__global__ void make_kdiff_kernel(const float* __restrict__ cw, float* __restrict__ kd) {
    int i = blockIdx.x * blockDim.x + threadIdx.x;
    if (i < 64 * 64) {
        float s = 0.f;
        for (int t = 0; t < 9; ++t) s += cw[(size_t)i * 9 + t];
        kd[i] = s;
    }
}

__global__ void make_scale_shift_kernel(const float* bias, const float* g, const float* be,
                                        const float* mu, const float* var,
                                        float* scale, float* shift) {
    int c = threadIdx.x;
    if (c < 64) {
        float sc = g[c] * rsqrtf(var[c] + 1e-5f);
        scale[c] = sc;
        shift[c] = (bias[c] - mu[c]) * sc + be[c];
    }
}

__global__ void cvt_bf16_kernel(const float* __restrict__ in,
                                unsigned short* __restrict__ out, int nelem) {
    int i = blockIdx.x * blockDim.x + threadIdx.x;
    if (i < nelem) out[i] = f2bf(in[i]);
}

__global__ void cvt_lidar_cat_kernel(const float* __restrict__ lidar,
                                     unsigned short* __restrict__ cat) {
    int i = blockIdx.x * blockDim.x + threadIdx.x;
    if (i >= NPIX) return;
    int w = i % WW;
    int h = (i / WW) % HH;
    int c = (i / (WW * HH)) % CC;
    int b = i / (WW * HH * CC);
    cat[(((size_t)b * 128 + 64 + c) * HH + h) * WW + w] = f2bf(lidar[i]);
}

// illumination gate: conv3x3 s2 (zero pad) + bias + BN + ReLU + GAP. Block=(b,oc).
__global__ void gate_kernel(const float* __restrict__ illu, const float* __restrict__ tw,
                            const float* __restrict__ tb, const float* __restrict__ g,
                            const float* __restrict__ be, const float* __restrict__ mu,
                            const float* __restrict__ var, float* __restrict__ gate) {
    __shared__ float red[256];
    int b = blockIdx.x / 9, oc = blockIdx.x % 9;
    float wv[9];
    for (int t2 = 0; t2 < 9; ++t2) wv[t2] = tw[oc * 9 + t2];
    float sc = g[oc] * rsqrtf(var[oc] + 1e-5f);
    float sh = tb[oc] * sc + be[oc] - mu[oc] * sc;
    const float* ip = illu + (size_t)b * (2 * HH) * (2 * WW);
    int t = threadIdx.x;
    float sum = 0.f;
    for (int p = t; p < HH * WW; p += 256) {
        int oh = p / WW, ow = p - oh * WW;
        float acc = 0.f;
        for (int ky = 0; ky < 3; ++ky) {
            int ih = 2 * oh + ky - 1;
            if (ih < 0 || ih >= 2 * HH) continue;
            for (int kx = 0; kx < 3; ++kx) {
                int iw = 2 * ow + kx - 1;
                if (iw < 0 || iw >= 2 * WW) continue;
                acc += wv[ky * 3 + kx] * ip[(size_t)ih * (2 * WW) + iw];
            }
        }
        sum += fmaxf(acc * sc + sh, 0.f);
    }
    red[t] = sum; __syncthreads();
    for (int s = 128; s > 0; s >>= 1) { if (t < s) red[t] += red[t + s]; __syncthreads(); }
    if (t == 0) gate[b * 9 + oc] = red[0] / (float)(HH * WW);
}

__global__ void norm_gate_kernel(const float* __restrict__ gate, float* __restrict__ sout) {
    __shared__ float sh[36];
    int t = threadIdx.x;
    if (t < 36) sh[t] = gate[t];
    __syncthreads();
    if (t < 36) {
        int b = t / 9;
        float d = 1e-8f;
        for (int n2 = 0; n2 < 9; ++n2) d += fabsf(sh[b * 9 + n2]);
        sout[t] = sh[t] / d;
    }
}

// weighted[b,c,h,w] = sum_n s[b,n] * out_normal[b,c,refl(h+dy),refl(w+dx)]  (reflect)
__global__ void weighted_kernel(const float* __restrict__ on, const float* __restrict__ sgate,
                                unsigned short* __restrict__ wout) {
    int i = blockIdx.x * blockDim.x + threadIdx.x;
    if (i >= NPIX) return;
    int w = i % WW;
    int h = (i / WW) % HH;
    int c = (i / (WW * HH)) % CC;
    int b = i / (WW * HH * CC);
    int hi[3] = { (h == 0) ? 1 : h - 1, h, (h == HH - 1) ? HH - 2 : h + 1 };
    int wi[3] = { (w == 0) ? 1 : w - 1, w, (w == WW - 1) ? WW - 2 : w + 1 };
    const float* base = on + ((size_t)b * CC + c) * HH * WW;
    float sv[9];
    for (int n2 = 0; n2 < 9; ++n2) sv[n2] = sgate[b * 9 + n2];
    float s = 0.f;
    for (int dy = 0; dy < 3; ++dy)
        for (int dx = 0; dx < 3; ++dx)
            s += sv[dy * 3 + dx] * base[(size_t)hi[dy] * WW + wi[dx]];
    wout[i] = f2bf(s);
}

// strip pools of x: xmW[b,c,h] = mean_w x, xmH[b,c,w] = mean_h x. Block=(b,c).
__global__ void pool_kernel(const float* __restrict__ x,
                            float* __restrict__ xmW, float* __restrict__ xmH) {
    int bc = blockIdx.x;
    const float* p = x + (size_t)bc * HH * WW;
    int t = threadIdx.x;
    for (int h = t; h < HH; h += blockDim.x) {
        float s = 0.f;
        for (int w = 0; w < WW; ++w) s += p[(size_t)h * WW + w];
        xmW[(size_t)bc * HH + h] = s * (1.0f / WW);
    }
    for (int w = t; w < WW; w += blockDim.x) {
        float s = 0.f;
        for (int h = 0; h < HH; ++h) s += p[(size_t)h * WW + w];
        xmH[(size_t)bc * WW + w] = s * (1.0f / HH);
    }
}

// q_pool = q_w @ xmW + q_b ; k_pool = k_w @ xmH + k_b   (pooling commutes with 1x1)
__global__ void qk_pool_kernel(const float* __restrict__ xmW, const float* __restrict__ xmH,
                               const float* __restrict__ qw, const float* __restrict__ qb,
                               const float* __restrict__ kw, const float* __restrict__ kb,
                               float* __restrict__ qpool, float* __restrict__ kpool) {
    int i = blockIdx.x * blockDim.x + threadIdx.x;
    const int NQ = BB * CC * HH;
    const int NK = BB * CC * WW;
    if (i < NQ) {
        int h = i % HH; int o = (i / HH) % CC; int b = i / (HH * CC);
        float s = qb[o];
        for (int c = 0; c < CC; ++c)
            s += qw[o * CC + c] * xmW[((size_t)b * CC + c) * HH + h];
        qpool[i] = s;
    } else if (i < NQ + NK) {
        int i2 = i - NQ;
        int w = i2 % WW; int o = (i2 / WW) % CC; int b = i2 / (WW * CC);
        float s = kb[o];
        for (int c = 0; c < CC; ++c)
            s += kw[o * CC + c] * xmH[((size_t)b * CC + c) * WW + w];
        kpool[i2] = s;
    }
}

// per-(b,c) softmax stats over rank-1 energy qp_h * kp_w (max via kmax/kmin trick)
__global__ void softmax_stats_kernel(const float* __restrict__ qpool,
                                     const float* __restrict__ kpool,
                                     float* __restrict__ mx, float* __restrict__ dn) {
    __shared__ float red[256];
    int bc = blockIdx.x;
    int t  = threadIdx.x;
    const float* qp = qpool + (size_t)bc * HH;
    const float* kp = kpool + (size_t)bc * WW;
    float km = -3.4e38f, kn = 3.4e38f;
    for (int w = t; w < WW; w += 256) { float v = kp[w]; km = fmaxf(km, v); kn = fminf(kn, v); }
    red[t] = km; __syncthreads();
    for (int s = 128; s > 0; s >>= 1) { if (t < s) red[t] = fmaxf(red[t], red[t + s]); __syncthreads(); }
    km = red[0]; __syncthreads();
    red[t] = kn; __syncthreads();
    for (int s = 128; s > 0; s >>= 1) { if (t < s) red[t] = fminf(red[t], red[t + s]); __syncthreads(); }
    kn = red[0]; __syncthreads();
    float m = -3.4e38f;
    for (int h = t; h < HH; h += 256) { float q = qp[h]; m = fmaxf(m, (q >= 0.f) ? q * km : q * kn); }
    red[t] = m; __syncthreads();
    for (int s = 128; s > 0; s >>= 1) { if (t < s) red[t] = fmaxf(red[t], red[t + s]); __syncthreads(); }
    m = red[0]; __syncthreads();
    float sum = 0.f;
    for (int i = t; i < HH * WW; i += 256) {
        int h = i / WW, w = i - h * WW;
        sum += __expf(qp[h] * kp[w] - m);
    }
    red[t] = sum; __syncthreads();
    for (int s = 128; s > 0; s >>= 1) { if (t < s) red[t] += red[t + s]; __syncthreads(); }
    if (t == 0) { mx[bc] = m; dn[bc] = red[0]; }
}

__global__ void final_kernel(const float* __restrict__ v, const float* __restrict__ x,
                             const float* __restrict__ qpool, const float* __restrict__ kpool,
                             const float* __restrict__ mx, const float* __restrict__ dn,
                             const float* __restrict__ gamma, float* __restrict__ out) {
    int i = blockIdx.x * blockDim.x + threadIdx.x;
    if (i >= NPIX) return;
    int w = i % WW;
    int h = (i / WW) % HH;
    int c = (i / (WW * HH)) % CC;
    int b = i / (WW * HH * CC);
    int bc = b * CC + c;
    float e = __expf(qpool[(size_t)bc * HH + h] * kpool[(size_t)bc * WW + w] - mx[bc]);
    out[i] = gamma[0] * v[i] * e / dn[bc] + x[i];
}

extern "C" void kernel_launch(void* const* d_in, const int* in_sizes, int n_in,
                              void* d_out, int out_size, void* d_ws, size_t ws_size,
                              hipStream_t stream) {
    (void)in_sizes; (void)n_in; (void)out_size; (void)ws_size;
    const float* lidar      = (const float*)d_in[0];
    const float* guidance   = (const float*)d_in[1];
    const float* illu       = (const float*)d_in[2];
    const float* conv_w     = (const float*)d_in[3];
    const float* transfer_w = (const float*)d_in[4];
    const float* transfer_b = (const float*)d_in[5];
    const float* bn_t_gamma = (const float*)d_in[6];
    const float* bn_t_beta  = (const float*)d_in[7];
    const float* bn_t_mean  = (const float*)d_in[8];
    const float* bn_t_var   = (const float*)d_in[9];
    const float* rgbd_w     = (const float*)d_in[10];
    const float* rgbd_b     = (const float*)d_in[11];
    const float* bn_r_gamma = (const float*)d_in[12];
    const float* bn_r_beta  = (const float*)d_in[13];
    const float* bn_r_mean  = (const float*)d_in[14];
    const float* bn_r_var   = (const float*)d_in[15];
    const float* q_w        = (const float*)d_in[16];
    const float* q_b        = (const float*)d_in[17];
    const float* k_w        = (const float*)d_in[18];
    const float* k_b        = (const float*)d_in[19];
    const float* v_w        = (const float*)d_in[20];
    const float* v_b        = (const float*)d_in[21];
    const float* pam_gamma  = (const float*)d_in[22];
    const float* theta      = (const float*)d_in[23];

    char* ws = (char*)d_ws;
    const size_t SZF = (size_t)NPIX * 4;   // 52,428,800
    const size_t SZB = (size_t)NPIX * 2;   // 26,214,400
    float*          out_normal = (float*)(ws);                         // reused as v
    unsigned short* g_bf       = (unsigned short*)(ws + SZF);          // reused as weighted
    unsigned short* cat_bf     = (unsigned short*)(ws + SZF + SZB);
    float*          xf         = (float*)(ws + SZF + SZB + SZF);
    unsigned short* x_bf       = (unsigned short*)(ws + 2 * SZF + SZB + SZF);
    char* sm = ws + 3 * SZF + 2 * SZB;
    float* kd        = (float*)(sm);                       // 16384
    float* scalebuf  = (float*)(sm + 16384);               // 256
    float* shiftbuf  = (float*)(sm + 16640);               // 256
    float* gatebuf   = (float*)(sm + 16896);               // 256
    float* sbuf      = (float*)(sm + 17152);               // 256
    float* xmW       = (float*)(sm + 17408);               // 163840
    float* xmH       = (float*)(sm + 181248);              // 327680
    float* qpool     = (float*)(sm + 508928);              // 163840
    float* kpool     = (float*)(sm + 672768);              // 327680
    float* mxbuf     = (float*)(sm + 1000448);             // 1024
    float* dnbuf     = (float*)(sm + 1001472);             // 1024
    unsigned int* frag_conv = (unsigned int*)(sm + 1002496); // 73728 (T=9,KC=2)
    unsigned int* frag_rgbd = (unsigned int*)(sm + 1076224); // 147456 (T=9,KC=4)
    unsigned int* frag_kd   = (unsigned int*)(sm + 1223680); // 8192   (T=1,KC=2)
    unsigned int* frag_v    = (unsigned int*)(sm + 1231872); // 8192   (T=1,KC=2)
    float*          vbuf    = out_normal;  // alias: out_normal dead before v GEMM
    unsigned short* weighted = g_bf;       // alias: g_bf dead before weighted write

    // --- prep ---
    make_kdiff_kernel<<<16, 256, 0, stream>>>(conv_w, kd);
    make_scale_shift_kernel<<<1, 64, 0, stream>>>(rgbd_b, bn_r_gamma, bn_r_beta,
                                                  bn_r_mean, bn_r_var, scalebuf, shiftbuf);
    frag_weights_kernel<<<72, 256, 0, stream>>>(conv_w, 64, 9, frag_conv, 18432);
    frag_weights_kernel<<<144, 256, 0, stream>>>(rgbd_w, 128, 9, frag_rgbd, 36864);
    frag_weights_kernel<<<8, 256, 0, stream>>>(kd, 64, 1, frag_kd, 2048);
    frag_weights_kernel<<<8, 256, 0, stream>>>(v_w, 64, 1, frag_v, 2048);
    cvt_bf16_kernel<<<(NPIX + 255) / 256, 256, 0, stream>>>(guidance, g_bf, NPIX);
    cvt_lidar_cat_kernel<<<(NPIX + 255) / 256, 256, 0, stream>>>(lidar, cat_bf);
    gate_kernel<<<36, 256, 0, stream>>>(illu, transfer_w, transfer_b, bn_t_gamma,
                                        bn_t_beta, bn_t_mean, bn_t_var, gatebuf);
    norm_gate_kernel<<<1, 64, 0, stream>>>(gatebuf, sbuf);

    // --- WMMA convs / GEMMs ---
    dim3 gconv(WW / 64, HH, BB);
    // out_normal = conv3x3(guidance, conv_w)
    conv_wmma_kernel<64, 9, 0><<<gconv, 128, 0, stream>>>(
        g_bf, frag_conv, nullptr, nullptr, out_normal, nullptr, 0);
    // weighted = sum_n s[b,n] * reflect-shift_n(out_normal)
    weighted_kernel<<<(NPIX + 255) / 256, 256, 0, stream>>>(out_normal, sbuf, weighted);
    // cat[:, :64] = bf16(out_normal - theta * kernel_diff @ weighted)
    conv_wmma_kernel<64, 1, 2><<<gconv, 128, 0, stream>>>(
        weighted, frag_kd, out_normal, theta, nullptr, cat_bf, 128);
    // x = relu(bn(conv3x3(cat, rgbd_w) + rgbd_b)) ; also bf16 copy for v GEMM
    conv_wmma_kernel<128, 9, 1><<<gconv, 128, 0, stream>>>(
        cat_bf, frag_rgbd, scalebuf, shiftbuf, xf, x_bf, 64);

    // --- PAM ---
    pool_kernel<<<BB * CC, 256, 0, stream>>>(xf, xmW, xmH);
    qk_pool_kernel<<<(BB * CC * (HH + WW) + 255) / 256, 256, 0, stream>>>(
        xmW, xmH, q_w, q_b, k_w, k_b, qpool, kpool);
    // v = v_w @ x + v_b  (full 1x1 WMMA GEMM)
    conv_wmma_kernel<64, 1, 3><<<gconv, 128, 0, stream>>>(
        x_bf, frag_v, v_b, nullptr, vbuf, nullptr, 0);
    softmax_stats_kernel<<<BB * CC, 256, 0, stream>>>(qpool, kpool, mxbuf, dnbuf);
    final_kernel<<<(NPIX + 255) / 256, 256, 0, stream>>>(
        vbuf, xf, qpool, kpool, mxbuf, dnbuf, pam_gamma, (float*)d_out);
}
